// Block_12945031430841
// MI455X (gfx1250) — matrix operand order repaired and checked by reading, not compiled, not run on hardware
//
#include <hip/hip_runtime.h>

// Problem constants (B, T, E, H from the reference)
#define E_DIM 1024
#define T_SEQ 2048
#define B_SZ  2
#define NH    16
#define DHD   64

typedef __attribute__((ext_vector_type(16))) __bf16 v16bf;
typedef __attribute__((ext_vector_type(8)))  float  v8f;

union FragU {
  v16bf v;
  unsigned int u[8];
  unsigned short h[16];
};

__device__ __forceinline__ unsigned short f2bf(float f) {
  union { __bf16 b; unsigned short s; } u;
  u.b = (__bf16)f;
  return u.s;
}

// A-matrix 16x32 bf16 fragment from row-major [M][K] memory (ld in elements).
// Lane L: M = L&15, half = L>>4. u[j] packs K = 16*(j/4) + 8*half + 2*(j%4) (+1).
__device__ __forceinline__ v16bf load_a_frag(const unsigned short* base, int ld,
                                             int m0, int k0, int lane) {
  const int half = lane >> 4, ml = lane & 15;
  const unsigned short* row = base + (size_t)(m0 + ml) * ld + k0;
  FragU f;
#pragma unroll
  for (int j = 0; j < 8; ++j) {
    const int k = ((j >> 2) << 4) + (half << 3) + ((j & 3) << 1);
    f.u[j] = *reinterpret_cast<const unsigned int*>(row + k);
  }
  return f.v;
}

// B-matrix 32x16 bf16 fragment from row-major [N][K] memory (i.e. B transposed
// in memory; ld in elements). Lane L: N = L&15, K = 16*(j/4) + 8*half + 2*(j%4) (+1).
__device__ __forceinline__ v16bf load_b_frag(const unsigned short* base, int ld,
                                             int n0, int k0, int lane) {
  const int half = lane >> 4, nl = lane & 15;
  const unsigned short* row = base + (size_t)(n0 + nl) * ld + k0 + (half << 3);
  FragU f;
#pragma unroll
  for (int j = 0; j < 8; ++j)
    f.u[j] = *reinterpret_cast<const unsigned int*>(row + ((j >> 2) << 4) + ((j & 3) << 1));
  return f.v;
}

// ---------------- prep kernels ----------------
__global__ void cvt_bf16_kernel(const float* __restrict__ src,
                                unsigned short* __restrict__ dst, int n) {
  const int i = blockIdx.x * blockDim.x + threadIdx.x;
  if (i < n) dst[i] = f2bf(src[i]);
}

// wt[n][k] = bf16(w[k][n])  (E x E)
__global__ void transpose_w_kernel(const float* __restrict__ w,
                                   unsigned short* __restrict__ wt) {
  const int k = blockIdx.x * 16 + threadIdx.x;
  const int n = blockIdx.y * 16 + threadIdx.y;
  wt[(size_t)n * E_DIM + k] = f2bf(w[(size_t)k * E_DIM + n]);
}

// ---------------- QKV projection ----------------
// One wave -> 16x64 output tile of one of {q, k, v}. q pre-scaled by 1/32.
// v is stored transposed per (b,h): vt[((b*NH+h)*DHD + d)*T + t].
// Double-buffered k-loop: fragments for k+32 load while k's WMMAs execute.
__global__ void gemm_qkv_kernel(const unsigned short* __restrict__ xb,
                                const unsigned short* __restrict__ wqt,
                                const unsigned short* __restrict__ wkt,
                                const unsigned short* __restrict__ wvt,
                                unsigned short* __restrict__ qb,
                                unsigned short* __restrict__ kb,
                                unsigned short* __restrict__ vt) {
  const int lane = threadIdx.x & 31;
  int w = (blockIdx.x * blockDim.x + threadIdx.x) >> 5;
  const int tilesM = (B_SZ * T_SEQ) / 16;   // 256
  const int tilesN = E_DIM / 64;            // 16
  const int out = w / (tilesM * tilesN);
  w -= out * tilesM * tilesN;
  const int mt = w / tilesN, nt = w % tilesN;
  const int m0 = mt * 16, n0 = nt * 64;
  const unsigned short* wsel = (out == 0) ? wqt : (out == 1) ? wkt : wvt;

  v8f acc[4] = {};
  v16bf a = load_a_frag(xb, E_DIM, m0, 0, lane);
  v16bf b[4];
#pragma unroll
  for (int i = 0; i < 4; ++i) b[i] = load_b_frag(wsel, E_DIM, n0 + i * 16, 0, lane);

  for (int k0 = 0; k0 < E_DIM; k0 += 32) {
    const int kn = (k0 + 32 < E_DIM) ? (k0 + 32) : 0;  // wrap: harmless reload
    v16bf an = load_a_frag(xb, E_DIM, m0, kn, lane);
    v16bf bn[4];
#pragma unroll
    for (int i = 0; i < 4; ++i) bn[i] = load_b_frag(wsel, E_DIM, n0 + i * 16, kn, lane);
#pragma unroll
    for (int i = 0; i < 4; ++i)
      acc[i] = __builtin_amdgcn_wmma_f32_16x16x32_bf16(
          false, a, false, b[i], (short)0, acc[i], false, false);
    a = an;
#pragma unroll
    for (int i = 0; i < 4; ++i) b[i] = bn[i];
  }

  const int half = lane >> 4, l16 = lane & 15;
#pragma unroll
  for (int i = 0; i < 4; ++i) {
#pragma unroll
    for (int r = 0; r < 8; ++r) {
      const int m = m0 + r + 8 * half;
      const int n = n0 + i * 16 + l16;
      const float v = acc[i][r];
      if (out == 0) {
        qb[(size_t)m * E_DIM + n] = f2bf(v * 0.03125f);  // fold 1/sqrt(E)
      } else if (out == 1) {
        kb[(size_t)m * E_DIM + n] = f2bf(v);
      } else {
        const int bb = m >> 11, tt = m & (T_SEQ - 1);
        const int hh = n >> 6, dd = n & (DHD - 1);
        vt[((size_t)(bb * NH + hh) * DHD + dd) * T_SEQ + tt] = f2bf(v);
      }
    }
  }
}

// ---------------- flash attention ----------------
// One wave per (b, h, 16-query tile). Online softmax over 32-key blocks.
__global__ void attn_kernel(const unsigned short* __restrict__ qb,
                            const unsigned short* __restrict__ kb,
                            const unsigned short* __restrict__ vt,
                            const float* __restrict__ x,
                            float* __restrict__ y,
                            unsigned short* __restrict__ yb) {
  __shared__ unsigned short Plds[4][16 * 32];  // per-wave P staging (bf16 bits)
  const int wid = threadIdx.x >> 5, lane = threadIdx.x & 31;
  const int half = lane >> 4, l16 = lane & 15;
  int gw = blockIdx.x * 4 + wid;               // 0 .. B*NH*(T/16)-1
  const int qt = gw & (T_SEQ / 16 - 1); gw >>= 7;
  const int h  = gw & (NH - 1);         gw >>= 4;
  const int b  = gw;
  const int q0 = qt * 16;

  const unsigned short* qbb = qb + (size_t)b * T_SEQ * E_DIM;
  const unsigned short* kbb = kb + (size_t)b * T_SEQ * E_DIM;
  const unsigned short* vth = vt + (size_t)(b * NH + h) * DHD * T_SEQ;

  const v16bf Qa0 = load_a_frag(qbb, E_DIM, q0, h * DHD,      lane);
  const v16bf Qa1 = load_a_frag(qbb, E_DIM, q0, h * DHD + 32, lane);

  float mrow[8], lrow[8];
#pragma unroll
  for (int r = 0; r < 8; ++r) { mrow[r] = -1.0e30f; lrow[r] = 0.0f; }
  v8f O[4] = {};
  unsigned short* Pl = &Plds[wid][0];

  for (int j0 = 0; j0 < q0 + 16; j0 += 32) {
    // S = (Q/32) @ K^T over dh=64 (two K-steps), two 16-key column tiles
    v8f S[2] = {};
#pragma unroll
    for (int nt = 0; nt < 2; ++nt) {
      v16bf bk0 = load_b_frag(kbb, E_DIM, j0 + nt * 16, h * DHD, lane);
      S[nt] = __builtin_amdgcn_wmma_f32_16x16x32_bf16(
          false, Qa0, false, bk0, (short)0, S[nt], false, false);
      v16bf bk1 = load_b_frag(kbb, E_DIM, j0 + nt * 16, h * DHD + 32, lane);
      S[nt] = __builtin_amdgcn_wmma_f32_16x16x32_bf16(
          false, Qa1, false, bk1, (short)0, S[nt], false, false);
    }
    // Issue V fragment loads now: their latency hides behind the softmax VALU
    // work and the P->LDS round-trip below.
    v16bf bv[4];
#pragma unroll
    for (int i = 0; i < 4; ++i)
      bv[i] = load_b_frag(vth, T_SEQ, i * 16, j0, lane);

    // causal mask
#pragma unroll
    for (int nt = 0; nt < 2; ++nt)
#pragma unroll
      for (int r = 0; r < 8; ++r) {
        const int row = q0 + r + 8 * half;
        const int col = j0 + nt * 16 + l16;
        const float sv = S[nt][r];
        S[nt][r] = (col > row) ? -1.0e30f : sv;
      }
    // online softmax statistics (rows live across 16-lane halves)
    float nm[8], sc[8];
#pragma unroll
    for (int r = 0; r < 8; ++r) {
      float mx = fmaxf(S[0][r], S[1][r]);
#pragma unroll
      for (int o = 1; o < 16; o <<= 1) mx = fmaxf(mx, __shfl_xor(mx, o, 32));
      nm[r] = fmaxf(mrow[r], mx);
      sc[r] = __expf(mrow[r] - nm[r]);
      mrow[r] = nm[r];
    }
#pragma unroll
    for (int nt = 0; nt < 2; ++nt)
#pragma unroll
      for (int r = 0; r < 8; ++r)
        S[nt][r] = __expf(S[nt][r] - nm[r]);
#pragma unroll
    for (int r = 0; r < 8; ++r) {
      float s = S[0][r] + S[1][r];
#pragma unroll
      for (int o = 1; o < 16; o <<= 1) s += __shfl_xor(s, o, 32);
      lrow[r] = lrow[r] * sc[r] + s;
    }
#pragma unroll
    for (int i = 0; i < 4; ++i)
#pragma unroll
      for (int r = 0; r < 8; ++r) O[i][r] *= sc[r];

    // P (C-layout) -> LDS row-major [16][32] -> A-fragment layout
#pragma unroll
    for (int nt = 0; nt < 2; ++nt)
#pragma unroll
      for (int r = 0; r < 8; ++r)
        Pl[(r + 8 * half) * 32 + nt * 16 + l16] = f2bf(S[nt][r]);
    FragU pf;
#pragma unroll
    for (int e = 0; e < 16; ++e) {
      const int k = ((e >> 3) << 4) + (half << 3) + (e & 7);
      pf.h[e] = Pl[l16 * 32 + k];
    }
    // O += P @ V  (V transposed: contiguous along tokens)
#pragma unroll
    for (int i = 0; i < 4; ++i)
      O[i] = __builtin_amdgcn_wmma_f32_16x16x32_bf16(
          false, pf.v, false, bv[i], (short)0, O[i], false, false);
  }
  // epilogue: normalize, residual with f32 x, emit y (f32) + y (bf16)
#pragma unroll
  for (int i = 0; i < 4; ++i)
#pragma unroll
    for (int r = 0; r < 8; ++r) {
      const int m = b * T_SEQ + q0 + r + 8 * half;
      const int n = h * DHD + i * 16 + l16;
      const float attnv = O[i][r] / lrow[r];
      const float yv = x[(size_t)m * E_DIM + n] + attnv;
      y[(size_t)m * E_DIM + n] = yv;
      yb[(size_t)m * E_DIM + n] = f2bf(yv);
    }
}

// ---------------- feed-forward ----------------
// out = y + relu(y @ Wf + bf);  one wave -> 16x64 tile, double-buffered k-loop
__global__ void ff_kernel(const unsigned short* __restrict__ yb,
                          const float* __restrict__ y,
                          const unsigned short* __restrict__ wft,
                          const float* __restrict__ bias,
                          float* __restrict__ out) {
  const int lane = threadIdx.x & 31;
  const int w = (blockIdx.x * blockDim.x + threadIdx.x) >> 5;
  const int tilesN = E_DIM / 64;  // 16
  const int mt = w / tilesN, nt = w % tilesN;
  const int m0 = mt * 16, n0 = nt * 64;

  v8f acc[4] = {};
  v16bf a = load_a_frag(yb, E_DIM, m0, 0, lane);
  v16bf b[4];
#pragma unroll
  for (int i = 0; i < 4; ++i) b[i] = load_b_frag(wft, E_DIM, n0 + i * 16, 0, lane);

  for (int k0 = 0; k0 < E_DIM; k0 += 32) {
    const int kn = (k0 + 32 < E_DIM) ? (k0 + 32) : 0;
    v16bf an = load_a_frag(yb, E_DIM, m0, kn, lane);
    v16bf bn[4];
#pragma unroll
    for (int i = 0; i < 4; ++i) bn[i] = load_b_frag(wft, E_DIM, n0 + i * 16, kn, lane);
#pragma unroll
    for (int i = 0; i < 4; ++i)
      acc[i] = __builtin_amdgcn_wmma_f32_16x16x32_bf16(
          false, a, false, b[i], (short)0, acc[i], false, false);
    a = an;
#pragma unroll
    for (int i = 0; i < 4; ++i) b[i] = bn[i];
  }

  const int half = lane >> 4, l16 = lane & 15;
#pragma unroll
  for (int i = 0; i < 4; ++i)
#pragma unroll
    for (int r = 0; r < 8; ++r) {
      const int m = m0 + r + 8 * half;
      const int n = n0 + i * 16 + l16;
      float v = acc[i][r] + bias[n];
      v = fmaxf(v, 0.0f);
      out[(size_t)m * E_DIM + n] = y[(size_t)m * E_DIM + n] + v;
    }
}

extern "C" void kernel_launch(void* const* d_in, const int* in_sizes, int n_in,
                              void* d_out, int out_size, void* d_ws, size_t ws_size,
                              hipStream_t stream) {
  const float* x  = (const float*)d_in[0];
  const float* Wq = (const float*)d_in[1];
  const float* Wk = (const float*)d_in[2];
  const float* Wv = (const float*)d_in[3];
  const float* Wf = (const float*)d_in[4];
  const float* bf = (const float*)d_in[5];
  float* out = (float*)d_out;

  char* ws = (char*)d_ws;
  const size_t MiB = 1024 * 1024;
  unsigned short* xb  = (unsigned short*)(ws);              //  8 MiB (bf16 x)
  unsigned short* wqt = (unsigned short*)(ws +  8 * MiB);   //  2 MiB each
  unsigned short* wkt = (unsigned short*)(ws + 10 * MiB);
  unsigned short* wvt = (unsigned short*)(ws + 12 * MiB);
  unsigned short* wft = (unsigned short*)(ws + 14 * MiB);
  unsigned short* qb  = (unsigned short*)(ws + 16 * MiB);   //  8 MiB
  unsigned short* kb  = (unsigned short*)(ws + 24 * MiB);   //  8 MiB
  unsigned short* vt  = (unsigned short*)(ws + 32 * MiB);   //  8 MiB
  float*          y   = (float*)        (ws + 40 * MiB);    // 16 MiB
  unsigned short* yb  = xb;  // xb is dead after gemm_qkv; reuse for bf16 y

  const int nx = B_SZ * T_SEQ * E_DIM;  // 4194304
  cvt_bf16_kernel<<<nx / 256, 256, 0, stream>>>(x, xb, nx);

  dim3 tb(16, 16), tg(E_DIM / 16, E_DIM / 16);
  transpose_w_kernel<<<tg, tb, 0, stream>>>(Wq, wqt);
  transpose_w_kernel<<<tg, tb, 0, stream>>>(Wk, wkt);
  transpose_w_kernel<<<tg, tb, 0, stream>>>(Wv, wvt);
  transpose_w_kernel<<<tg, tb, 0, stream>>>(Wf, wft);

  // 3 outputs * 256 M-tiles * 16 N64-tiles = 12288 waves; 8 waves/block
  gemm_qkv_kernel<<<1536, 256, 0, stream>>>(xb, wqt, wkt, wvt, qb, kb, vt);

  // B*NH*(T/16) = 4096 waves; 4 waves/block
  attn_kernel<<<1024, 128, 0, stream>>>(qb, kb, vt, x, y, yb);

  // 256 * 16 = 4096 waves; 8 waves/block
  ff_kernel<<<512, 256, 0, stream>>>(yb, y, wft, bf, out);
}
